// ResidualMapNet_61598420959439
// MI455X (gfx1250) — matrix operand order repaired and checked by reading, not compiled, not run on hardware
//
#include <hip/hip_runtime.h>
#include <hip/hip_bf16.h>

// Fused ResidualMapNet for MI455X (gfx1250, wave32, WMMA), v3.
//
// v3 vs v2 (driven by disasm: per-block W2/W1 LDS staging cost as many DS ops
// as the whole compute phase, duplicated across 14400 blocks):
//  - Weights pre-packed ONCE into exact WMMA fragment order in d_ws
//    (zeros baked in) -> fused kernel loads fragments with coalesced
//    global_load_b64/b128 from a 20KB L2-hot buffer. No LDS weight staging,
//    no __syncthreads, no exec-masked fragment branches.
//  - LDS = wave-private hidden tile only (33.8KB/block, barrier-free).
//  - feats transposed to (N,HiWi,C) -> per-pixel 32-channel gather is one
//    128B line, lanes 0-15 read 64B coalesced (L2-resident either way).
//  - Fallback (tiny ws) keeps the v2 LDS-staging scheme.

typedef __attribute__((ext_vector_type(16))) _Float16 v16h;
typedef __attribute__((ext_vector_type(8)))  _Float16 v8h;
typedef __attribute__((ext_vector_type(4)))  _Float16 v4h;
typedef __attribute__((ext_vector_type(8)))  float    v8f;
typedef __attribute__((ext_vector_type(4)))  float    v4f;

#define NB   4
#define CCH  32
#define HI   240
#define WI   240
#define HO   480
#define WO   480
#define HIWI (HI*WI)     // 57600
#define HOWO (HO*WO)     // 230400
#define HID  256
#define HPAD 264         // 528B row = 132 dwords == 4 (mod 64 banks) -> conflict-free

__device__ __forceinline__ float half_max16(float v) {
#pragma unroll
  for (int m = 1; m < 16; m <<= 1) {
    float o = __shfl_xor(v, m, 32);
    v = fmaxf(v, o);
  }
  return v;
}
__device__ __forceinline__ float half_sum16(float v) {
#pragma unroll
  for (int m = 1; m < 16; m <<= 1) v += __shfl_xor(v, m, 32);
  return v;
}
__device__ __forceinline__ v16h cat8(v8h lo, v8h hi) {
  return __builtin_shufflevector(lo, hi, 0, 1, 2, 3, 4, 5, 6, 7,
                                 8, 9, 10, 11, 12, 13, 14, 15);
}

// bias + relu + row softmax (half-wave shfl_xor) + gather + weighted sum.
// out[p] = sum_c feats[c]*lw[c] + depth   (softmax rows sum to 1)
template <bool TR>
__device__ __forceinline__ void finish_tile(
    const v8f& acc0, const v8f& acc1, int tileBase, int hb, int ln,
    const int* __restrict__ mapping, const float* __restrict__ fsrc,
    const float* __restrict__ depth, const float* __restrict__ b2,
    float* __restrict__ out)
{
  const float bias0 = b2[ln];
  const float bias1 = b2[16 + ln];
  float lw0[8], lw1[8];
#pragma unroll
  for (int r = 0; r < 8; ++r) {
    float v0 = acc0[r] + bias0; v0 = v0 > 0.f ? v0 : 0.f;
    float v1 = acc1[r] + bias1; v1 = v1 > 0.f ? v1 : 0.f;
    const float mx = half_max16(fmaxf(v0, v1));
    const float e0 = __expf(v0 - mx);
    const float e1 = __expf(v1 - mx);
    const float s  = half_sum16(e0 + e1);
    const float inv = 1.0f / s;
    lw0[r] = e0 * inv;
    lw1[r] = e1 * inv;
  }
#pragma unroll
  for (int r = 0; r < 8; ++r) {
    const int p = tileBase + r + hb;
    const int n = p / HOWO;
    const int q = p - n * HOWO;
    const int m0 = mapping[2 * q];
    const int m1 = mapping[2 * q + 1];
    const int sp = m0 * WI + m1;
    float f0, f1;
    if (TR) {
      const size_t base = ((size_t)n * HIWI + (size_t)sp) * CCH;
      f0 = fsrc[base + ln];          // lanes 0-15: 64B coalesced
      f1 = fsrc[base + 16 + ln];
    } else {
      const size_t base = (size_t)n * CCH * HIWI + (size_t)sp;
      f0 = fsrc[base + (size_t)ln * HIWI];
      f1 = fsrc[base + (size_t)(ln + 16) * HIWI];
    }
    const float tot = half_sum16(f0 * lw0[r] + f1 * lw1[r]);
    if (ln == 0) out[p] = tot + depth[(size_t)n * HIWI + sp];
  }
}

// ---------------- FAST path: fragment-packed weights in global ----------------
__global__ __launch_bounds__(128) void residual_mapnet_fast(
    const float*    __restrict__ pos,     // (N,Ho,Wo,3)
    const int*      __restrict__ mapping, // (Ho,Wo,2)
    const float*    __restrict__ featsT,  // (N,Hi*Wi,C)
    const float*    __restrict__ depth,   // (N,1,Hi,Wi)
    const _Float16* __restrict__ PW1,     // [nc][lane][4]  A-frag packed
    const float*    __restrict__ b1,      // (256,)
    const _Float16* __restrict__ PW2,     // [kc*2+ncu][lane][16] B-frag packed
    const float*    __restrict__ b2,      // (32,)
    float*          __restrict__ out)     // (N,1,Ho,Wo)
{
  __shared__ __align__(16) _Float16 sH[4][16 * HPAD];  // wave-private hidden tile

  const int tid  = threadIdx.x;
  const int wave = tid >> 5;
  const int lane = tid & 31;
  const int ln   = lane & 15;
  const bool hiH = lane >= 16;
  const int hb   = hiH ? 8 : 0;

  const int tileBase = (blockIdx.x * 4 + wave) * 16;
  _Float16* myH = sH[wave];

  // pos as B fragment (K(e)=e+(hi?16:0); only K=0..2 valid -> low lanes).
  // Branchless: all lanes load (hi lanes redundantly), then mask.
  const float* pp = pos + (size_t)(tileBase + ln) * 3;
  const float p0 = pp[0], p1 = pp[1], p2 = pp[2];
  v16h bp = {};
  bp[0] = hiH ? (_Float16)0.f : (_Float16)p0;
  bp[1] = hiH ? (_Float16)0.f : (_Float16)p1;
  bp[2] = hiH ? (_Float16)0.f : (_Float16)p2;

  // ---- Layer 1 (transposed): one WMMA + one b64 weight load + one b128
  // store per 16-unit chunk; zeros for hi lanes are baked into PW1.
  for (int nc = 0; nc < 16; ++nc) {
    const v4h w4 = *(const v4h*)&PW1[(nc * 32 + lane) * 4];  // coalesced 8B/lane
    v16h aw = {};
    aw[0] = w4[0]; aw[1] = w4[1]; aw[2] = w4[2]; aw[3] = w4[3];
    v8f z = {};
    v8f d = __builtin_amdgcn_wmma_f32_16x16x32_f16(
        false, aw, false, bp, (short)0, z, false, false);
    const v4f bA = *(const v4f*)&b1[nc * 16 + hb];       // broadcast within half
    const v4f bB = *(const v4f*)&b1[nc * 16 + hb + 4];
    v8h hp;
#pragma unroll
    for (int r = 0; r < 8; ++r) {
      float h = d[r] + (r < 4 ? bA[r] : bB[r - 4]);
      h = h > 0.f ? h : 0.f;
      hp[r] = (_Float16)h;
    }
    *(v8h*)&myH[ln * HPAD + nc * 16 + hb] = hp;          // 16B aligned, conflict-free
  }

  // ---- Layer 2: A from LDS (2x b128), B from packed global (2x2 b128) ----
  v8f acc0 = {}, acc1 = {};
  for (int kc = 0; kc < 8; ++kc) {
    const _Float16* hrow = &myH[ln * HPAD + kc * 32];
    const v16h af = cat8(*(const v8h*)(hrow + 2 * hb), *(const v8h*)(hrow + 2 * hb + 8));
    const _Float16* q0 = &PW2[(size_t)((kc * 2 + 0) * 32 + lane) * 16];  // coalesced
    const _Float16* q1 = &PW2[(size_t)((kc * 2 + 1) * 32 + lane) * 16];
    const v16h bf0 = cat8(*(const v8h*)q0, *(const v8h*)(q0 + 8));
    const v16h bf1 = cat8(*(const v8h*)q1, *(const v8h*)(q1 + 8));
    acc0 = __builtin_amdgcn_wmma_f32_16x16x32_f16(
        false, af, false, bf0, (short)0, acc0, false, false);
    acc1 = __builtin_amdgcn_wmma_f32_16x16x32_f16(
        false, af, false, bf1, (short)0, acc1, false, false);
  }

  finish_tile<true>(acc0, acc1, tileBase, hb, ln, mapping, featsT, depth, b2, out);
}

// ---------------- Fallback: LDS-staged weights, direct gather ----------------
__global__ __launch_bounds__(128) void residual_mapnet_basic(
    const float* __restrict__ pos, const int* __restrict__ mapping,
    const float* __restrict__ feats, const float* __restrict__ depth,
    const float* __restrict__ W1, const float* __restrict__ b1,
    const float* __restrict__ W2, const float* __restrict__ b2,
    float* __restrict__ out)
{
  __shared__ __align__(16) _Float16 sW1T[HID * 4];     // [n][k], k=0..2 valid
  __shared__ __align__(16) _Float16 sW2T[CCH * HPAD];  // [n][k] transposed
  __shared__ __align__(16) _Float16 sH[4][16 * HPAD];

  const int tid  = threadIdx.x;
  const int wave = tid >> 5;
  const int lane = tid & 31;
  const int ln   = lane & 15;
  const bool hiH = lane >= 16;
  const int hb   = hiH ? 8 : 0;
  const int kbB  = hiH ? 16 : 0;

  for (int i = tid; i < 3 * HID; i += 128) {
    const int k = i / HID, n = i - k * HID;
    sW1T[n * 4 + k] = (_Float16)W1[i];
  }
  for (int i = tid; i < HID * CCH; i += 128) {
    const int k = i / CCH, n = i - k * CCH;
    sW2T[n * HPAD + k] = (_Float16)W2[i];
  }
  __syncthreads();

  const int tileBase = (blockIdx.x * 4 + wave) * 16;
  _Float16* myH = sH[wave];

  const float* pp = pos + (size_t)(tileBase + ln) * 3;
  const float p0 = pp[0], p1 = pp[1], p2 = pp[2];
  v16h bp = {};
  bp[0] = hiH ? (_Float16)0.f : (_Float16)p0;
  bp[1] = hiH ? (_Float16)0.f : (_Float16)p1;
  bp[2] = hiH ? (_Float16)0.f : (_Float16)p2;

  for (int nc = 0; nc < 16; ++nc) {
    v16h aw = {};
    if (!hiH) {
      const _Float16* w = &sW1T[(nc * 16 + ln) * 4];
      aw[0] = w[0]; aw[1] = w[1]; aw[2] = w[2];
    }
    v8f z = {};
    v8f d = __builtin_amdgcn_wmma_f32_16x16x32_f16(
        false, aw, false, bp, (short)0, z, false, false);
    const v4f bA = *(const v4f*)&b1[nc * 16 + hb];
    const v4f bB = *(const v4f*)&b1[nc * 16 + hb + 4];
    v8h hp;
#pragma unroll
    for (int r = 0; r < 8; ++r) {
      float h = d[r] + (r < 4 ? bA[r] : bB[r - 4]);
      h = h > 0.f ? h : 0.f;
      hp[r] = (_Float16)h;
    }
    *(v8h*)&myH[ln * HPAD + nc * 16 + hb] = hp;
  }

  v8f acc0 = {}, acc1 = {};
  for (int kc = 0; kc < 8; ++kc) {
    const _Float16* hrow = &myH[ln * HPAD + kc * 32];
    const v16h af = cat8(*(const v8h*)(hrow + 2 * hb), *(const v8h*)(hrow + 2 * hb + 8));
    const _Float16* w0 = &sW2T[ln * HPAD + kc * 32 + kbB];
    const _Float16* w1 = &sW2T[(ln + 16) * HPAD + kc * 32 + kbB];
    const v16h bf0 = cat8(*(const v8h*)w0, *(const v8h*)(w0 + 8));
    const v16h bf1 = cat8(*(const v8h*)w1, *(const v8h*)(w1 + 8));
    acc0 = __builtin_amdgcn_wmma_f32_16x16x32_f16(
        false, af, false, bf0, (short)0, acc0, false, false);
    acc1 = __builtin_amdgcn_wmma_f32_16x16x32_f16(
        false, af, false, bf1, (short)0, acc1, false, false);
  }

  finish_tile<false>(acc0, acc1, tileBase, hb, ln, mapping, feats, depth, b2, out);
}

// ---------------- setup kernels ----------------
__global__ __launch_bounds__(256) void transpose_feats(
    const float* __restrict__ feats, float* __restrict__ featsT) {
  const int i = blockIdx.x * 256 + threadIdx.x;  // over N*C*HIWI, coalesced read
  const int sp = i % HIWI;
  const int t  = i / HIWI;
  const int c  = t % CCH;
  const int n  = t / CCH;
  featsT[((size_t)n * HIWI + sp) * CCH + c] = feats[i];
}

// Pack W1/W2 into exact per-lane WMMA fragment order (zeros baked in).
__global__ __launch_bounds__(256) void pack_weights(
    const float* __restrict__ W1, const float* __restrict__ W2,
    _Float16* __restrict__ PW1, _Float16* __restrict__ PW2) {
  const int tid = threadIdx.x;  // single block
  // PW1[nc][lane][e]: A-frag, K(e)=e (low lanes), valid K=0..2
  for (int i = tid; i < 16 * 32 * 4; i += 256) {
    const int nc = i >> 7, lane = (i >> 2) & 31, e = i & 3;
    float v = 0.f;
    if (lane < 16 && e < 3) v = W1[e * HID + nc * 16 + lane];
    PW1[i] = (_Float16)v;
  }
  // PW2[kc*2+ncu][lane][e]: B-frag, K = kc*32 + (lane>=16?16:0) + e, N = ncu*16 + lane%16
  for (int i = tid; i < 16 * 32 * 16; i += 256) {
    const int kcn = i >> 9, lane = (i >> 4) & 31, e = i & 15;
    const int kc = kcn >> 1, ncu = kcn & 1;
    const int K = kc * 32 + (lane >= 16 ? 16 : 0) + e;
    const int N = ncu * 16 + (lane & 15);
    PW2[i] = (_Float16)W2[K * CCH + N];
  }
}

extern "C" void kernel_launch(void* const* d_in, const int* in_sizes, int n_in,
                              void* d_out, int out_size, void* d_ws, size_t ws_size,
                              hipStream_t stream) {
  const float* pos     = (const float*)d_in[0];
  const int*   mapping = (const int*)  d_in[1];
  const float* feats   = (const float*)d_in[2];
  const float* depth   = (const float*)d_in[3];
  const float* W1      = (const float*)d_in[4];
  const float* b1      = (const float*)d_in[5];
  const float* W2      = (const float*)d_in[6];
  const float* b2      = (const float*)d_in[7];
  float* out = (float*)d_out;

  const int totalPixels = NB * HOWO;      // 921600
  const int blocks = totalPixels / 64;    // 4 waves x 16 pixels per block

  const size_t featsTBytes = (size_t)NB * HIWI * CCH * sizeof(float);  // 29,491,200
  const size_t pw1Bytes = (size_t)16 * 32 * 4 * sizeof(_Float16);      // 4 KB
  const size_t pw2Bytes = (size_t)16 * 32 * 16 * sizeof(_Float16);     // 16 KB

  if (ws_size >= featsTBytes + pw1Bytes + pw2Bytes) {
    float*    featsT = (float*)d_ws;
    _Float16* PW1    = (_Float16*)((char*)d_ws + featsTBytes);
    _Float16* PW2    = (_Float16*)((char*)d_ws + featsTBytes + pw1Bytes);
    const int nElem = NB * CCH * HIWI;    // 7,372,800
    transpose_feats<<<nElem / 256, 256, 0, stream>>>(feats, featsT);
    pack_weights<<<1, 256, 0, stream>>>(W1, W2, PW1, PW2);
    residual_mapnet_fast<<<blocks, 128, 0, stream>>>(
        pos, mapping, featsT, depth, PW1, b1, PW2, b2, out);
  } else {
    residual_mapnet_basic<<<blocks, 128, 0, stream>>>(
        pos, mapping, feats, depth, W1, b1, W2, b2, out);
  }
}
